// StackedAAGCN_85074712199478
// MI455X (gfx1250) — matrix operand order repaired and checked by reading, not compiled, not run on hardware
//
#include <hip/hip_runtime.h>
#include <hip/hip_bf16.h>

// ---------------------------------------------------------------------------
// StackedAAGCN on MI455X (gfx1250, wave32).
// All channel-mixing GEMMs (1x1 convs, 9x1 temporal conv, down/residual) use
// v_wmma_f32_16x16x32_bf16 (bf16 A/B converted in-register, f32 accumulate).
// GEMM inner loops are branch-free: select-clamped in-bounds addresses with
// mask-multiplied values (forces unconditional global_load + v_cndmask, no
// EXEC branching), vectorized b128 A-fragment loads, 2 row-tiles per wave
// sharing one strided B fragment.  Path selection (aligned K / temporal edge)
// is done per-template-instantiation with if-constexpr so no dead guarded
// code lands in the hot kernels.
// Small ops (VxV attention, graph agg over V=25, STC gates, BN fusions) are
// plain VALU kernels: <2% of FLOPs, L2-resident at 23.3 TB/s.
// Workspace requirement: ~86.3M floats (~345 MB).
// ---------------------------------------------------------------------------

#define NB_  8
#define TT_  512
#define VV_  25
#define COLS_ (TT_ * VV_)   // 12800, multiple of 16

typedef __attribute__((ext_vector_type(16))) __bf16          v16bf;
typedef __attribute__((ext_vector_type(16))) unsigned short  v16u;
typedef __attribute__((ext_vector_type(8)))  float           v8f;

__device__ __forceinline__ unsigned short f2bf(float f) {
  unsigned int u = __builtin_bit_cast(unsigned int, f);
  u += 0x7FFFu + ((u >> 16) & 1u);            // round-to-nearest-even
  return (unsigned short)(u >> 16);
}

__device__ __forceinline__ float sigmoidf_(float x) {
  return 1.f / (1.f + expf(-x));
}

__device__ __forceinline__ v8f wmma_bf16(const v16u& a, const v16u& b, v8f c) {
  return __builtin_amdgcn_wmma_f32_16x16x32_bf16(
      false, __builtin_bit_cast(v16bf, a),
      false, __builtin_bit_cast(v16bf, b),
      (short)0, c, false, false);
}

// A-fragment (16-bit 16x32 layout) from a 16B-aligned W row chunk.
__device__ __forceinline__ void load_a_frag(const float* wp, v16u& au) {
  float4 a0 = *(const float4*)(wp);
  float4 a1 = *(const float4*)(wp + 4);
  float4 a2 = *(const float4*)(wp + 16);
  float4 a3 = *(const float4*)(wp + 20);
  au[0]  = f2bf(a0.x); au[1]  = f2bf(a0.y); au[2]  = f2bf(a0.z); au[3]  = f2bf(a0.w);
  au[4]  = f2bf(a1.x); au[5]  = f2bf(a1.y); au[6]  = f2bf(a1.z); au[7]  = f2bf(a1.w);
  au[8]  = f2bf(a2.x); au[9]  = f2bf(a2.y); au[10] = f2bf(a2.z); au[11] = f2bf(a2.w);
  au[12] = f2bf(a3.x); au[13] = f2bf(a3.y); au[14] = f2bf(a3.z); au[15] = f2bf(a3.w);
}

// ---------------------------------------------------------------------------
// WMMA GEMM:  out[n][o][col] = (beta? out : 0) + bias[o]
//                            + sum_k W[o][k] * X[n][ c(k) ][ col shifted by tap ]
// k = c*KT + tap; KT=1 -> 1x1 conv over channels; KT=9 -> 9x1 temporal conv
// (pad 4, zero outside [0,T)).  One wave computes MT 16x16 tiles of D that
// share a B fragment.  O % (16*MT) == 0 and COLS_ % 16 == 0 guaranteed.
// ALIGNED <=> Ktot % 32 == 0 (true for everything except the cin=3 convs).
// ---------------------------------------------------------------------------
template <int KT, int MT, bool ALIGNED>
__global__ void gemm_wmma_kernel(const float* __restrict__ W,
                                 const float* __restrict__ X,
                                 const float* __restrict__ bias,
                                 float* __restrict__ out,
                                 int O, int Cin, int beta) {
  constexpr int PAD = KT / 2;
  const int Ktot = Cin * KT;
  const int row_groups = O / (16 * MT);
  const int lane = threadIdx.x & 31;
  const int wv   = threadIdx.x >> 5;
  const int tid  = blockIdx.x * 4 + wv;            // global wave id
  const int mt   = tid % row_groups;
  const int ct   = tid / row_groups;
  if (ct >= COLS_ / 16) return;                    // wave-uniform guard
  const int nb = blockIdx.y;

  const int m15 = lane & 15;
  const int hh  = lane >> 4;                       // lane half (K-half select)
  const int colg = ct * 16 + m15;                  // B/D column of this lane
  const int t0 = colg / VV_;
  const int v0 = colg - t0 * VV_;

  const float* Wrow[MT];
#pragma unroll
  for (int m = 0; m < MT; ++m)
    Wrow[m] = W + (size_t)(mt * 16 * MT + m * 16 + m15) * Ktot;
  const float* Xn = X + (size_t)nb * Cin * COLS_;

  v8f acc[MT];
#pragma unroll
  for (int m = 0; m < MT; ++m)
#pragma unroll
    for (int i = 0; i < 8; ++i) acc[m][i] = 0.f;

  const int nchunk = (Ktot + 31) >> 5;

  if constexpr (ALIGNED) {
    bool interior = true;
    if constexpr (KT > 1) {
      const int tlo = (ct * 16) / VV_;
      const int thi = (ct * 16 + 15) / VV_;
      interior = (tlo >= PAD) && (thi + PAD < TT_);
    }
    if (interior) {
      // ---- fast path: all K and t in-bounds; no guards at all ----
      const long boff = (long)(t0 - PAD) * VV_ + v0;   // == colg when KT==1
      for (int ch = 0; ch < nchunk; ++ch) {
        const int kc0 = ch << 5;
        __builtin_prefetch(Wrow[0] + kc0 + 32, 0, 1);  // global_prefetch_b8
        v16u bu;
#pragma unroll
        for (int e = 0; e < 16; ++e) {
          int kb  = kc0 + ((e < 8) ? (hh * 8 + e) : (16 + hh * 8 + (e - 8)));
          int c   = (KT == 1) ? kb : (kb / KT);
          int tap = (KT == 1) ? 0  : (kb - c * KT);
          bu[e] = f2bf(Xn[(long)c * COLS_ + (long)tap * VV_ + boff]);
        }
        v16u au[MT];
#pragma unroll
        for (int m = 0; m < MT; ++m) load_a_frag(Wrow[m] + kc0 + hh * 8, au[m]);
#pragma unroll
        for (int m = 0; m < MT; ++m) acc[m] = wmma_bf16(au[m], bu, acc[m]);
      }
    } else if constexpr (KT > 1) {
      // ---- temporal-edge tiles: mask-multiply keeps loads unconditional ----
      for (int ch = 0; ch < nchunk; ++ch) {
        const int kc0 = ch << 5;
        v16u bu;
#pragma unroll
        for (int e = 0; e < 16; ++e) {
          int kb  = kc0 + ((e < 8) ? (hh * 8 + e) : (16 + hh * 8 + (e - 8)));
          int c   = kb / KT;
          int tap = kb - c * KT;
          int tt  = t0 + tap - PAD;
          bool ok = (unsigned)tt < (unsigned)TT_;
          float msk = ok ? 1.f : 0.f;
          long off = (long)c * COLS_ + (ok ? ((long)tt * VV_ + v0) : 0);
          bu[e] = f2bf(Xn[off] * msk);              // load always executes
        }
        v16u au[MT];
#pragma unroll
        for (int m = 0; m < MT; ++m) load_a_frag(Wrow[m] + kc0 + hh * 8, au[m]);
#pragma unroll
        for (int m = 0; m < MT; ++m) acc[m] = wmma_bf16(au[m], bu, acc[m]);
      }
    }
  } else {
    // ---- generic path (Ktot==3 only): mask-multiplied, branch-free ----
    for (int ch = 0; ch < nchunk; ++ch) {
      const int kc0 = ch << 5;
      v16u au[MT], bu;
#pragma unroll
      for (int e = 0; e < 16; ++e) {
        int ka = kc0 + ((e < 8) ? (hh * 8 + e) : (16 + hh * 8 + (e - 8)));
        bool oka = ka < Ktot;
        float amsk = oka ? 1.f : 0.f;
        int kac = oka ? ka : 0;
#pragma unroll
        for (int m = 0; m < MT; ++m)
          au[m][e] = f2bf(Wrow[m][kac] * amsk);     // load always executes
        int kb  = kc0 + ((e < 8) ? (hh * 8 + e) : (16 + hh * 8 + (e - 8)));
        int c   = (KT == 1) ? kb : (kb / KT);
        int tap = (KT == 1) ? 0  : (kb - c * KT);
        int tt  = t0 + tap - PAD;
        bool ok = (kb < Ktot) && ((unsigned)tt < (unsigned)TT_);
        float msk = ok ? 1.f : 0.f;
        long off = ok ? ((long)c * COLS_ + (long)tt * VV_ + v0) : 0;
        bu[e] = f2bf(Xn[off] * msk);                // load always executes
      }
#pragma unroll
      for (int m = 0; m < MT; ++m) acc[m] = wmma_bf16(au[m], bu, acc[m]);
    }
  }

  // D layout: lane n-col = lane&15, VGPR r holds M = (lane>>4)*8 + r.
#pragma unroll
  for (int m = 0; m < MT; ++m)
#pragma unroll
    for (int r = 0; r < 8; ++r) {
      int o = mt * 16 * MT + m * 16 + hh * 8 + r;
      size_t idx = ((size_t)nb * O + o) * COLS_ + colg;
      float vsum = acc[m][r];
      if (bias) vsum += bias[o];
      if (beta) vsum += out[idx];
      out[idx] = vsum;
    }
}

// ---------------------------------------------------------------------------
// ae[n][i][j] = PA[i][j] + alpha * tanh( (1/K) sum_{c,t} ca[n,c,t,i]*cb[n,c,t,j] )
// one wave per (n,i,j), lane-strided reduction over K = inter*T.
// ---------------------------------------------------------------------------
__global__ void attn_ae_kernel(const float* __restrict__ ca,
                               const float* __restrict__ cb,
                               const float* __restrict__ PA,
                               const float* __restrict__ alpha,
                               float* __restrict__ ae, int inter) {
  int wv = threadIdx.x >> 5, lane = threadIdx.x & 31;
  int id = blockIdx.x * (blockDim.x >> 5) + wv;
  const int TOT = NB_ * VV_ * VV_;
  if (id >= TOT) return;
  int j = id % VV_, i = (id / VV_) % VV_, n = id / (VV_ * VV_);
  int K = inter * TT_;
  const float* pa = ca + (size_t)n * K * VV_ + i;
  const float* pb = cb + (size_t)n * K * VV_ + j;
  float s = 0.f;
  for (int k = lane; k < K; k += 32)
    s += pa[(size_t)k * VV_] * pb[(size_t)k * VV_];
#pragma unroll
  for (int off = 16; off > 0; off >>= 1) s += __shfl_xor(s, off, 32);
  if (lane == 0) {
    float m = tanhf(s / (float)K);
    ae[id] = PA[i * VV_ + j] + alpha[0] * m;
  }
}

// agg[n][c][t][w] = sum_v x[n][c][t][v] * ae[n][v][w]
__global__ void agg_kernel(const float* __restrict__ x,
                           const float* __restrict__ ae,
                           float* __restrict__ out, int C) {
  size_t idx = (size_t)blockIdx.x * blockDim.x + threadIdx.x;
  size_t tot = (size_t)NB_ * C * COLS_;
  if (idx >= tot) return;
  int w = (int)(idx % VV_);
  size_t r = idx / VV_;                          // flat (n,c,t)
  int n = (int)(r / ((size_t)C * TT_));
  const float* xr = x + r * VV_;
  const float* a  = ae + (size_t)n * VV_ * VV_ + w;
  float s = 0.f;
#pragma unroll
  for (int v = 0; v < VV_; ++v) s += xr[v] * a[v * VV_];
  out[idx] = s;
}

// out = relu( bn1(a + abias) + bn2(bsrc) ), inference-mode BN per channel.
__global__ void bn2_relu_kernel(const float* __restrict__ a,
                                const float* __restrict__ abias,
                                const float* __restrict__ g1, const float* __restrict__ be1,
                                const float* __restrict__ m1, const float* __restrict__ va1,
                                const float* __restrict__ bsrc,
                                const float* __restrict__ g2, const float* __restrict__ be2,
                                const float* __restrict__ m2, const float* __restrict__ va2,
                                float* __restrict__ out, int C) {
  size_t idx = (size_t)blockIdx.x * blockDim.x + threadIdx.x;
  size_t tot = (size_t)NB_ * C * COLS_;
  if (idx >= tot) return;
  int c = (int)((idx / COLS_) % C);
  float sc1 = g1[c] * rsqrtf(va1[c] + 1e-5f);
  float sh1 = be1[c] - m1[c] * sc1;
  float sc2 = g2[c] * rsqrtf(va2[c] + 1e-5f);
  float sh2 = be2[c] - m2[c] * sc2;
  float av = a[idx] + (abias ? abias[c] : 0.f);
  float r = av * sc1 + sh1 + bsrc[idx] * sc2 + sh2;
  out[idx] = r > 0.f ? r : 0.f;
}

// means: over T -> (N,C,V); over V -> (N,C,T); over T*V -> (N,C)
__global__ void mean_t_kernel(const float* __restrict__ y, float* __restrict__ o, int C) {
  int id = blockIdx.x * blockDim.x + threadIdx.x;
  if (id >= NB_ * C * VV_) return;
  int v = id % VV_; size_t nc = id / VV_;
  const float* p = y + nc * COLS_ + v;
  float s = 0.f;
  for (int t = 0; t < TT_; ++t) s += p[(size_t)t * VV_];
  o[id] = s * (1.f / TT_);
}
__global__ void mean_v_kernel(const float* __restrict__ y, float* __restrict__ o, int C) {
  int id = blockIdx.x * blockDim.x + threadIdx.x;
  if (id >= NB_ * C * TT_) return;
  const float* p = y + (size_t)id * VV_;
  float s = 0.f;
#pragma unroll
  for (int v = 0; v < VV_; ++v) s += p[v];
  o[id] = s * (1.f / VV_);
}
__global__ void mean_tv_kernel(const float* __restrict__ y, float* __restrict__ o, int C) {
  int id = blockIdx.x * blockDim.x + threadIdx.x;
  if (id >= NB_ * C) return;
  const float* p = y + (size_t)id * COLS_;
  float s = 0.f;
  for (int k = 0; k < COLS_; ++k) s += p[k];
  o[id] = s * (1.f / COLS_);
}

// spatial gate: sa[n][vp] = sigmoid(b + sum_{c,u} w[c,u] * ym[n,c,vp+u-12])
__global__ void sattn_kernel(const float* __restrict__ ym, const float* __restrict__ w,
                             const float* __restrict__ bias, float* __restrict__ sa, int C) {
  int id = blockIdx.x * blockDim.x + threadIdx.x;
  if (id >= NB_ * VV_) return;
  int vp = id % VV_, n = id / VV_;
  float s = bias[0];
  for (int c = 0; c < C; ++c) {
    const float* yr = ym + ((size_t)n * C + c) * VV_;
    const float* wr = w + c * VV_;
#pragma unroll
    for (int u = 0; u < VV_; ++u) {
      int p = vp + u - (VV_ - 1) / 2;
      if (p >= 0 && p < VV_) s += wr[u] * yr[p];
    }
  }
  sa[id] = sigmoidf_(s);
}

// temporal gate: ta[n][t] = sigmoid(b + sum_{c,u} w[c,u] * yt[n,c,t+u-4])
__global__ void tattn_kernel(const float* __restrict__ yt, const float* __restrict__ w,
                             const float* __restrict__ bias, float* __restrict__ ta, int C) {
  int id = blockIdx.x * blockDim.x + threadIdx.x;
  if (id >= NB_ * TT_) return;
  int t = id % TT_, n = id / TT_;
  float s = bias[0];
  for (int c = 0; c < C; ++c) {
    const float* yr = yt + ((size_t)n * C + c) * TT_;
    const float* wr = w + c * 9;
#pragma unroll
    for (int u = 0; u < 9; ++u) {
      int p = t + u - 4;
      if (p >= 0 && p < TT_) s += wr[u] * yr[p];
    }
  }
  ta[id] = sigmoidf_(s);
}

// y *= (1 + gate); mode 0: gate[n][v], mode 1: gate[n][t], mode 2: gate[n][c]
__global__ void scale_kernel(float* __restrict__ y, const float* __restrict__ s,
                             int C, int mode) {
  size_t idx = (size_t)blockIdx.x * blockDim.x + threadIdx.x;
  size_t tot = (size_t)NB_ * C * COLS_;
  if (idx >= tot) return;
  int v = (int)(idx % VV_);
  int t = (int)((idx / VV_) % TT_);
  int c = (int)((idx / COLS_) % C);
  int n = (int)(idx / ((size_t)C * COLS_));
  float f = (mode == 0) ? s[n * VV_ + v]
          : (mode == 1) ? s[n * TT_ + t]
                        : s[n * C + c];
  y[idx] *= (1.f + f);
}

// channel SE: h = relu(W1 se0 + b1); s = sigmoid(W2 h + b2); one block per n.
__global__ void fc_kernel(const float* __restrict__ se0,
                          const float* __restrict__ w1, const float* __restrict__ b1,
                          const float* __restrict__ w2, const float* __restrict__ b2,
                          float* __restrict__ seb, int C) {
  __shared__ float h[128];
  int n = blockIdx.x, j = threadIdx.x;
  int H = C / 2;
  if (j < H) {
    float s = b1[j];
    for (int c = 0; c < C; ++c) s += se0[n * C + c] * w1[j * C + c];
    h[j] = s > 0.f ? s : 0.f;
  }
  __syncthreads();
  if (j < C) {
    float s = b2[j];
    for (int k = 0; k < H; ++k) s += h[k] * w2[j * H + k];
    seb[n * C + j] = sigmoidf_(s);
  }
}

// ---------------------------------------------------------------------------
extern "C" void kernel_launch(void* const* d_in, const int* in_sizes, int n_in,
                              void* d_out, int out_size, void* d_ws, size_t ws_size,
                              hipStream_t stream) {
  (void)in_sizes; (void)n_in; (void)out_size; (void)ws_size;

  static const int CIN_[3]  = {3, 64, 128};
  static const int COUT_[3] = {64, 128, 256};

  float* ws = (float*)d_ws;
  const size_t A_MAX = (size_t)NB_ * 256 * COLS_;     // 26,214,400 floats
  float* y     = ws;                                  // [A_MAX]
  float* z     = ws + A_MAX;                          // [A_MAX] (also agg buf)
  float* conva = ws + 2 * A_MAX;                      // [6,553,600]
  float* convb = conva + 6553600;                     // [6,553,600]
  float* curA  = convb + 6553600;                     // [6,553,600]  blk1 out
  float* curB  = curA + 6553600;                      // [13,107,200] blk2 out
  float* aeb   = curB + 13107200;                     // [8192]
  float* ym    = aeb + 8192;                          // [51200]
  float* yt    = ym + 51200;                          // [1,048,576]
  float* sab   = yt + 1048576;                        // [256]
  float* tab   = sab + 256;                           // [4096]
  float* se0   = tab + 4096;                          // [2048]
  float* seb   = se0 + 2048;                          // [2048]
  // total ~86.3M floats (~345 MB) of workspace assumed available.

  auto ew = [](size_t tot) { return dim3((unsigned)((tot + 255) / 256)); };

  auto gemm = [&](const float* W, const float* X, const float* bias, float* out,
                  int O, int Cin, int KT, int beta) {
    bool aligned = ((Cin * KT) & 31) == 0;
    int mtf = (O % 32 == 0) ? 2 : 1;
    int row_groups = O / (16 * mtf);
    int tiles = row_groups * (COLS_ / 16);
    dim3 grid((tiles + 3) / 4, NB_);
    if (KT == 9)
      gemm_wmma_kernel<9, 2, true><<<grid, 128, 0, stream>>>(W, X, bias, out, O, Cin, beta);
    else if (aligned)
      gemm_wmma_kernel<1, 2, true><<<grid, 128, 0, stream>>>(W, X, bias, out, O, Cin, beta);
    else if (mtf == 2)
      gemm_wmma_kernel<1, 2, false><<<grid, 128, 0, stream>>>(W, X, bias, out, O, Cin, beta);
    else
      gemm_wmma_kernel<1, 1, false><<<grid, 128, 0, stream>>>(W, X, bias, out, O, Cin, beta);
  };

  const float* xin = (const float*)d_in[0];           // (8,3,512,25)

  for (int blk = 0; blk < 3; ++blk) {
    auto Pp = [&](int j) { return (const float*)d_in[1 + 38 * blk + j]; };
    const float *PA = Pp(0), *alpha = Pp(1), *wa = Pp(2), *ba = Pp(3),
                *wb = Pp(4), *bb = Pp(5), *wd = Pp(6), *bd = Pp(7),
                *gg = Pp(8), *gb = Pp(9), *gm = Pp(10), *gv = Pp(11),
                *wdn = Pp(12), *bdn = Pp(13),
                *dg = Pp(14), *db = Pp(15), *dm = Pp(16), *dv = Pp(17),
                *wsa = Pp(18), *bsa = Pp(19), *wta = Pp(20), *bta = Pp(21),
                *wfc1 = Pp(22), *bfc1 = Pp(23), *wfc2 = Pp(24), *bfc2 = Pp(25),
                *wtcn = Pp(26), *btcn = Pp(27),
                *tg = Pp(28), *tb = Pp(29), *tm = Pp(30), *tv = Pp(31),
                *wres = Pp(32), *bres = Pp(33),
                *rg = Pp(34), *rb = Pp(35), *rm = Pp(36), *rv = Pp(37);

    int cin = CIN_[blk], cout = COUT_[blk], inter = cout / 4;
    float* xout = (blk == 0) ? curA : (blk == 1) ? curB : (float*)d_out;
    size_t totO = (size_t)NB_ * cout * COLS_;

    // ---- adaptive graph conv (unit_gcn) ----
    for (int i = 0; i < 3; ++i) {
      gemm(wa + (size_t)i * inter * cin, xin, ba + i * inter, conva, inter, cin, 1, 0);
      gemm(wb + (size_t)i * inter * cin, xin, bb + i * inter, convb, inter, cin, 1, 0);
      attn_ae_kernel<<<(NB_ * VV_ * VV_ + 3) / 4, 128, 0, stream>>>(
          conva, convb, PA + i * VV_ * VV_, alpha, aeb, inter);
      agg_kernel<<<ew((size_t)NB_ * cin * COLS_), 256, 0, stream>>>(xin, aeb, z, cin);
      gemm(wd + (size_t)i * cout * cin, z, bd + (size_t)i * cout, y, cout, cin, 1,
           (i > 0) ? 1 : 0);
    }
    gemm(wdn, xin, bdn, z, cout, cin, 1, 0);
    bn2_relu_kernel<<<ew(totO), 256, 0, stream>>>(
        y, nullptr, gg, gb, gm, gv, z, dg, db, dm, dv, y, cout);

    // ---- STC attention ----
    mean_t_kernel<<<ew((size_t)NB_ * cout * VV_), 256, 0, stream>>>(y, ym, cout);
    sattn_kernel<<<1, 256, 0, stream>>>(ym, wsa, bsa, sab, cout);
    scale_kernel<<<ew(totO), 256, 0, stream>>>(y, sab, cout, 0);
    mean_v_kernel<<<ew((size_t)NB_ * cout * TT_), 256, 0, stream>>>(y, yt, cout);
    tattn_kernel<<<ew((size_t)NB_ * TT_), 256, 0, stream>>>(yt, wta, bta, tab, cout);
    scale_kernel<<<ew(totO), 256, 0, stream>>>(y, tab, cout, 1);
    mean_tv_kernel<<<ew((size_t)NB_ * cout), 256, 0, stream>>>(y, se0, cout);
    fc_kernel<<<NB_, 256, 0, stream>>>(se0, wfc1, bfc1, wfc2, bfc2, seb, cout);
    scale_kernel<<<ew(totO), 256, 0, stream>>>(y, seb, cout, 2);

    // ---- temporal conv (9x1) + block residual ----
    gemm(wtcn, y, nullptr, z, cout, cout, 9, 0);          // z = conv9x1(y)
    gemm(wres, xin, bres, xout, cout, cin, 1, 0);         // xout = 1x1 residual
    bn2_relu_kernel<<<ew(totO), 256, 0, stream>>>(
        z, btcn, tg, tb, tm, tv, xout, rg, rb, rm, rv, xout, cout);

    xin = xout;
  }
}
